// MultiHeadSelfAttention_37263136260135
// MI455X (gfx1250) — matrix operand order repaired and checked by reading, not compiled, not run on hardware
//
#include <hip/hip_runtime.h>
#include <hip/hip_bf16.h>

typedef _Float16 v8h  __attribute__((ext_vector_type(8)));
typedef _Float16 v16h __attribute__((ext_vector_type(16)));
typedef float    v8f  __attribute__((ext_vector_type(8)));

#define EMBED 768
#define NH    12
#define HD    64
#define TSEQ  4096
#define BATCH 2
#define MTOT  (BATCH*TSEQ)   /* 8192 tokens */
#define N_QKV (3*EMBED)      /* 2304 */

// ---------- WMMA operand loaders (v_wmma_f32_16x16x32_f16 layouts) ----------
// A (16x32 f16): lane L holds row M=L&15; element e <-> K = 16*(L>>4) + e
//   => one contiguous 16-half run at K offset 16*half.
// B (32x16 f16) given as B^T row-major [N][K]: lane L holds col N=L&15;
//   element e <-> K = 16*(L>>4) + e  => one contiguous 16-half run.

__device__ __forceinline__ v16h h16cat(v8h lo, v8h hi) {
    return __builtin_shufflevector(lo, hi, 0,1,2,3,4,5,6,7,8,9,10,11,12,13,14,15);
}

__device__ __forceinline__ v16h load_a16(const _Float16* base, int ld, int lane) {
    int row  = lane & 15;
    int half = lane >> 4;
    const _Float16* p = base + (size_t)row * ld + 16 * half;
    v8h lo = *(const v8h*)(p);
    v8h hi = *(const v8h*)(p + 8);
    return h16cat(lo, hi);
}

__device__ __forceinline__ v16h load_bt16(const _Float16* bt, int ld, int lane) {
    int n    = lane & 15;
    int half = lane >> 4;
    const _Float16* p = bt + (size_t)n * ld + 16 * half;
    v8h lo = *(const v8h*)(p);
    v8h hi = *(const v8h*)(p + 8);
    return h16cat(lo, hi);
}

__device__ __forceinline__ v8f wmma16(v16h a, v16h b, v8f c) {
    return __builtin_amdgcn_wmma_f32_16x16x32_f16(false, a, false, b, (short)0, c, false, false);
}

// ---------- prep kernels ----------
__global__ void k_cvt(const float* __restrict__ src, _Float16* __restrict__ dst, int n) {
    int i = blockIdx.x * blockDim.x + threadIdx.x;
    if (i < n) dst[i] = (_Float16)src[i];
}

// w: [K][N] fp32 row-major  ->  wt: [N][K] f16 row-major (B^T layout)
__global__ void k_tcvt(const float* __restrict__ w, _Float16* __restrict__ wt, int K, int N) {
    int n = blockIdx.x * 16 + threadIdx.x;
    int k = blockIdx.y * 16 + threadIdx.y;
    if (n < N && k < K) wt[(size_t)n * K + k] = (_Float16)w[(size_t)k * N + n];
}

// ---------- QKV projection: 2x4 register-blocked (32x64 out per wave)
__global__ void __launch_bounds__(256)
k_qkv(const _Float16* __restrict__ xh, const _Float16* __restrict__ wt,
      const float* __restrict__ bias,
      _Float16* __restrict__ Qb, _Float16* __restrict__ Kb, _Float16* __restrict__ Vt) {
    int lane = threadIdx.x & 31;
    int wave = threadIdx.x >> 5;
    int wm = wave & 1, wn = wave >> 1;          // 2x4 wave grid
    int m0 = blockIdx.x * 64 + wm * 32;         // token tile (32 rows)
    int n0 = blockIdx.y * 256 + wn * 64;        // channel tile (64 cols)

    v8f vzero = {};
    v8f acc[2][4];
#pragma unroll
    for (int i = 0; i < 2; ++i)
#pragma unroll
        for (int j = 0; j < 4; ++j) acc[i][j] = vzero;

    for (int k0 = 0; k0 < EMBED; k0 += 32) {
        v16h a0 = load_a16(xh + (size_t)m0 * EMBED + k0,        EMBED, lane);
        v16h a1 = load_a16(xh + (size_t)(m0 + 16) * EMBED + k0, EMBED, lane);
        v16h b0 = load_bt16(wt + (size_t)(n0 +  0) * EMBED + k0, EMBED, lane);
        v16h b1 = load_bt16(wt + (size_t)(n0 + 16) * EMBED + k0, EMBED, lane);
        v16h b2 = load_bt16(wt + (size_t)(n0 + 32) * EMBED + k0, EMBED, lane);
        v16h b3 = load_bt16(wt + (size_t)(n0 + 48) * EMBED + k0, EMBED, lane);
        acc[0][0] = wmma16(a0, b0, acc[0][0]);
        acc[0][1] = wmma16(a0, b1, acc[0][1]);
        acc[0][2] = wmma16(a0, b2, acc[0][2]);
        acc[0][3] = wmma16(a0, b3, acc[0][3]);
        acc[1][0] = wmma16(a1, b0, acc[1][0]);
        acc[1][1] = wmma16(a1, b1, acc[1][1]);
        acc[1][2] = wmma16(a1, b2, acc[1][2]);
        acc[1][3] = wmma16(a1, b3, acc[1][3]);
    }

    int nn   = lane & 15;
    int half = lane >> 4;
#pragma unroll
    for (int j = 0; j < 4; ++j) {
        int n   = n0 + j * 16 + nn;
        float bv = bias[n];
        int sel = n / EMBED;            // 0=Q 1=K 2=V
        int c   = n - sel * EMBED;
        int h   = c >> 6, d = c & 63;
#pragma unroll
        for (int i = 0; i < 2; ++i) {
#pragma unroll
            for (int r = 0; r < 8; ++r) {
                int m  = m0 + i * 16 + r + 8 * half;   // global token id
                int bb = m >> 12;                       // T = 4096
                int t  = m & (TSEQ - 1);
                _Float16 hv = (_Float16)(acc[i][j][r] + bv);
                size_t bhh = (size_t)bb * NH + h;
                if (sel == 0)      Qb[(bhh * TSEQ + t) * HD + d] = hv;
                else if (sel == 1) Kb[(bhh * TSEQ + t) * HD + d] = hv;
                else               Vt[(bhh * HD + d) * TSEQ + t] = hv;
            }
        }
    }
}

// ---------- flash attention: 32 queries per wave, 32-key chunks, causal
#define QW 4   // waves per block
__global__ void __launch_bounds__(32 * QW)
k_attn(const _Float16* __restrict__ Qb, const _Float16* __restrict__ Kb,
       const _Float16* __restrict__ Vt, _Float16* __restrict__ Oh) {
    __shared__ __align__(16) _Float16 lds[QW][2 * 16 * 32];   // per-wave P transpose buffer
    int lane = threadIdx.x & 31;
    int wave = threadIdx.x >> 5;
    int half = lane >> 4;
    int nn   = lane & 15;

    int bh = blockIdx.y;                               // b*NH + h
    int q0 = (blockIdx.x * QW + wave) * 32;            // 32-query tile start
    const _Float16* Qh = Qb + (size_t)bh * TSEQ * HD;
    const _Float16* Kh = Kb + (size_t)bh * TSEQ * HD;
    const _Float16* Vh = Vt + (size_t)bh * HD * TSEQ;
    _Float16* pl = lds[wave];

    // Q tile (32 x 64): 2 query subtiles x 2 K-dim halves
    v16h aq[2][2];
#pragma unroll
    for (int i = 0; i < 2; ++i) {
        aq[i][0] = load_a16(Qh + (size_t)(q0 + i * 16) * HD + 0,  HD, lane);
        aq[i][1] = load_a16(Qh + (size_t)(q0 + i * 16) * HD + 32, HD, lane);
    }

    v8f vzero = {};
    float mrow[2][8], lrow[2][8];
    v8f O[2][4];
#pragma unroll
    for (int i = 0; i < 2; ++i) {
#pragma unroll
        for (int r = 0; r < 8; ++r) { mrow[i][r] = -1e30f; lrow[i][r] = 0.0f; }
#pragma unroll
        for (int j = 0; j < 4; ++j) O[i][j] = vzero;
    }

    const float scale = 0.125f;   // 1/sqrt(64)

    for (int kc = 0; kc < q0 + 32; kc += 32) {
        // ---- shared B operands for this chunk
        v16h kb[4];
        kb[0] = load_bt16(Kh + (size_t)kc * HD + 0,         HD, lane);  // keys kc..+15, d 0..31
        kb[1] = load_bt16(Kh + (size_t)(kc + 16) * HD + 0,  HD, lane);
        kb[2] = load_bt16(Kh + (size_t)kc * HD + 32,        HD, lane);
        kb[3] = load_bt16(Kh + (size_t)(kc + 16) * HD + 32, HD, lane);
        v16h vb[4];
#pragma unroll
        for (int j = 0; j < 4; ++j)
            vb[j] = load_bt16(Vh + (size_t)(j * 16) * TSEQ + kc, TSEQ, lane);

#pragma unroll
        for (int i = 0; i < 2; ++i) {
            // S = Q K^T (16 queries x 32 keys)
            v8f S0 = vzero, S1 = vzero;
            S0 = wmma16(aq[i][0], kb[0], S0);
            S0 = wmma16(aq[i][1], kb[2], S0);
            S1 = wmma16(aq[i][0], kb[1], S1);
            S1 = wmma16(aq[i][1], kb[3], S1);

            int qbase = q0 + i * 16;
            bool needMask = (kc + 31 > qbase);
            float rmax[8];
#pragma unroll
            for (int r = 0; r < 8; ++r) {
                float s0 = S0[r] * scale;
                float s1 = S1[r] * scale;
                if (needMask) {
                    int qg = qbase + r + 8 * half;
                    if (kc + nn      > qg) s0 = -1e30f;
                    if (kc + 16 + nn > qg) s1 = -1e30f;
                }
                S0[r] = s0; S1[r] = s1;
                float t = fmaxf(s0, s1);           // row lives in 16 lanes of this half
                t = fmaxf(t, __shfl_xor(t, 1, 32));
                t = fmaxf(t, __shfl_xor(t, 2, 32));
                t = fmaxf(t, __shfl_xor(t, 4, 32));
                t = fmaxf(t, __shfl_xor(t, 8, 32));
                rmax[r] = t;
            }
#pragma unroll
            for (int r = 0; r < 8; ++r) {
                float mnew  = fmaxf(mrow[i][r], rmax[r]);
                float alpha = __expf(mrow[i][r] - mnew);
                mrow[i][r] = mnew;
                float p0 = __expf(S0[r] - mnew);
                float p1 = __expf(S1[r] - mnew);
                float rs = p0 + p1;
                rs += __shfl_xor(rs, 1, 32);
                rs += __shfl_xor(rs, 2, 32);
                rs += __shfl_xor(rs, 4, 32);
                rs += __shfl_xor(rs, 8, 32);
                lrow[i][r] = lrow[i][r] * alpha + rs;
                O[i][0][r] *= alpha; O[i][1][r] *= alpha;
                O[i][2][r] *= alpha; O[i][3][r] *= alpha;
                int m = r + 8 * half;
                pl[i * 512 + m * 32 + nn]      = (_Float16)p0;
                pl[i * 512 + m * 32 + 16 + nn] = (_Float16)p1;
            }
        }
        asm volatile("s_wait_dscnt 0" ::: "memory");   // LDS writes -> reads, same wave

        // ---- O += P (16x32) x V (32x64) for both query subtiles
#pragma unroll
        for (int i = 0; i < 2; ++i) {
            v16h ap = load_a16(pl + i * 512, 32, lane);
            O[i][0] = wmma16(ap, vb[0], O[i][0]);
            O[i][1] = wmma16(ap, vb[1], O[i][1]);
            O[i][2] = wmma16(ap, vb[2], O[i][2]);
            O[i][3] = wmma16(ap, vb[3], O[i][3]);
        }
    }

    // ---- normalize and emit to [B*T][C] f16 (A operand of final projection)
    int b = bh / NH, h = bh % NH;
#pragma unroll
    for (int i = 0; i < 2; ++i) {
#pragma unroll
        for (int r = 0; r < 8; ++r) {
            float rl = 1.0f / lrow[i][r];
            int m = q0 + i * 16 + r + 8 * half;
            size_t rowoff = ((size_t)b * TSEQ + m) * EMBED + (size_t)h * HD;
            Oh[rowoff +  0 + nn] = (_Float16)(O[i][0][r] * rl);
            Oh[rowoff + 16 + nn] = (_Float16)(O[i][1][r] * rl);
            Oh[rowoff + 32 + nn] = (_Float16)(O[i][2][r] * rl);
            Oh[rowoff + 48 + nn] = (_Float16)(O[i][3][r] * rl);
        }
    }
}

// ---------- output projection: 2x4 register-blocked, fp32 result
__global__ void __launch_bounds__(256)
k_proj(const _Float16* __restrict__ ah, const _Float16* __restrict__ wt,
       const float* __restrict__ bias, float* __restrict__ out) {
    int lane = threadIdx.x & 31;
    int wave = threadIdx.x >> 5;
    int wm = wave & 1, wn = wave >> 1;
    int m0 = blockIdx.x * 64 + wm * 32;
    int n0 = blockIdx.y * 256 + wn * 64;

    v8f vzero = {};
    v8f acc[2][4];
#pragma unroll
    for (int i = 0; i < 2; ++i)
#pragma unroll
        for (int j = 0; j < 4; ++j) acc[i][j] = vzero;

    for (int k0 = 0; k0 < EMBED; k0 += 32) {
        v16h a0 = load_a16(ah + (size_t)m0 * EMBED + k0,        EMBED, lane);
        v16h a1 = load_a16(ah + (size_t)(m0 + 16) * EMBED + k0, EMBED, lane);
        v16h b0 = load_bt16(wt + (size_t)(n0 +  0) * EMBED + k0, EMBED, lane);
        v16h b1 = load_bt16(wt + (size_t)(n0 + 16) * EMBED + k0, EMBED, lane);
        v16h b2 = load_bt16(wt + (size_t)(n0 + 32) * EMBED + k0, EMBED, lane);
        v16h b3 = load_bt16(wt + (size_t)(n0 + 48) * EMBED + k0, EMBED, lane);
        acc[0][0] = wmma16(a0, b0, acc[0][0]);
        acc[0][1] = wmma16(a0, b1, acc[0][1]);
        acc[0][2] = wmma16(a0, b2, acc[0][2]);
        acc[0][3] = wmma16(a0, b3, acc[0][3]);
        acc[1][0] = wmma16(a1, b0, acc[1][0]);
        acc[1][1] = wmma16(a1, b1, acc[1][1]);
        acc[1][2] = wmma16(a1, b2, acc[1][2]);
        acc[1][3] = wmma16(a1, b3, acc[1][3]);
    }

    int nn   = lane & 15;
    int half = lane >> 4;
#pragma unroll
    for (int j = 0; j < 4; ++j) {
        int n = n0 + j * 16 + nn;
        float bv = bias[n];
#pragma unroll
        for (int i = 0; i < 2; ++i) {
#pragma unroll
            for (int r = 0; r < 8; ++r) {
                int m = m0 + i * 16 + r + 8 * half;
                out[(size_t)m * EMBED + n] = acc[i][j][r] + bv;
            }
        }
    }
}

extern "C" void kernel_launch(void* const* d_in, const int* in_sizes, int n_in,
                              void* d_out, int out_size, void* d_ws, size_t ws_size,
                              hipStream_t stream) {
    const float* x     = (const float*)d_in[0];
    const float* Wqkv  = (const float*)d_in[1];
    const float* bqkv  = (const float*)d_in[2];
    const float* Wproj = (const float*)d_in[3];
    const float* bproj = (const float*)d_in[4];
    float* out = (float*)d_out;

    // workspace carve-up (~55 MB; attn output aliases x_h, which is dead by then)
    char* ws = (char*)d_ws;
    size_t off = 0;
    auto carve = [&](size_t bytes) {
        char* p = ws + off;
        off += (bytes + 255) & ~(size_t)255;
        return p;
    };
    _Float16* xh  = (_Float16*)carve((size_t)MTOT * EMBED * 2);
    _Float16* wtq = (_Float16*)carve((size_t)N_QKV * EMBED * 2);
    _Float16* wtp = (_Float16*)carve((size_t)EMBED * EMBED * 2);
    _Float16* Qb  = (_Float16*)carve((size_t)MTOT * EMBED * 2);
    _Float16* Kb  = (_Float16*)carve((size_t)MTOT * EMBED * 2);
    _Float16* Vt  = (_Float16*)carve((size_t)MTOT * EMBED * 2);
    _Float16* Ah  = xh;   // reuse: xh only needed by k_qkv

    int n = MTOT * EMBED;
    k_cvt<<<(n + 255) / 256, 256, 0, stream>>>(x, xh, n);
    dim3 tb(16, 16);
    k_tcvt<<<dim3(N_QKV / 16, EMBED / 16), tb, 0, stream>>>(Wqkv, wtq, EMBED, N_QKV);
    k_tcvt<<<dim3(EMBED / 16, EMBED / 16), tb, 0, stream>>>(Wproj, wtp, EMBED, EMBED);

    k_qkv <<<dim3(MTOT / 64, N_QKV / 256), 256, 0, stream>>>(xh, wtq, bqkv, Qb, Kb, Vt);
    k_attn<<<dim3(TSEQ / 32 / QW, BATCH * NH), 32 * QW, 0, stream>>>(Qb, Kb, Vt, Ah);
    k_proj<<<dim3(MTOT / 64, EMBED / 256), 256, 0, stream>>>(Ah, wtp, bproj, out);
}
